// VQINR_26938034881058
// MI455X (gfx1250) — matrix operand order repaired and checked by reading, not compiled, not run on hardware
//
#include <hip/hip_runtime.h>
#include <hip/hip_bf16.h>

// ---------------- problem constants ----------------
#define BB   65536      // batch points
#define CC   2          // coords
#define VV   3          // output values
#define DD   64         // latent dim
#define KK   1024       // codebook entries
#define SS   4          // residual stages
#define HH   1024       // hidden width
#define LL   5          // modulated layers
#define FF   25         // frequencies
#define INR  102        // real PE width  = C*(2F+1)
#define INP  128        // padded PE width (K multiple of 32)

#define ROWS 64         // rows (points) per block in the MLP kernel
#define AST  1032       // LDS activation row stride in bf16 (1024 + 8 pad -> conflict-free b128)

typedef __attribute__((ext_vector_type(16))) __bf16 bfrag;   // 32B WMMA operand
typedef __attribute__((ext_vector_type(8)))  float  v8f;     // 16x16 f32 accum (8 VGPR)

union FragU { float4 h[2]; bfrag v; };

// ---------------- workspace layout ----------------
// [0, 20480)              : bias_total (L*H f32) = beta + dec bias, fused
// [32768, +262144)        : W0 fragments (1024 N x 128 K bf16, lane-swizzled)
// [294912, +8388608)      : Wh fragments (4 x 1024 x 1024 bf16, lane-swizzled)
#define WS_BIAS_OFF  0
#define WS_W0_OFF    32768
#define WS_WH_OFF    294912

// =====================================================================
// Kernel 1: residual VQ (forward), commit loss, fused bias_total
// =====================================================================
__global__ __launch_bounds__(256) void vq_prep(
    const float* __restrict__ latents, const int* __restrict__ latent_idx,
    const float* __restrict__ codebooks,
    const float* __restrict__ mod_W, const float* __restrict__ mod_b,
    const float* __restrict__ dec_b0, const float* __restrict__ dec_bh,
    float* __restrict__ bias_total, float* __restrict__ out_tail)
{
  __shared__ float zs[SS][DD];
  __shared__ float zsum[DD];
  __shared__ float red_d[256];
  __shared__ int   red_i[256];
  __shared__ float stage_loss[SS];

  const int t  = threadIdx.x;
  const int li = latent_idx[0];
  const float* img = latents + (long)li * SS * DD;

  if (t < SS * DD) zs[t / DD][t % DD] = img[t];
  __syncthreads();
  // forward value of z_q_sum is just the sum of stage latents (STE identity)
  if (t < DD) zsum[t] = zs[0][t] + zs[1][t] + zs[2][t] + zs[3][t];
  __syncthreads();

  for (int s = 0; s < SS; ++s) {
    const float* cb = codebooks + (long)s * KK * DD;
    float best_d = 3.402823466e38f; int best_i = 0;
    for (int j = 0; j < KK / 256; ++j) {
      const int k = t + j * 256;                 // ascending -> strict < keeps first min
      const float* e = cb + (long)k * DD;
      float d = 0.f;
      #pragma unroll 8
      for (int q = 0; q < DD; ++q) { float df = zs[s][q] - e[q]; d += df * df; }
      if (d < best_d) { best_d = d; best_i = k; }
    }
    red_d[t] = best_d; red_i[t] = best_i;
    __syncthreads();
    for (int off = 128; off > 0; off >>= 1) {
      if (t < off) {
        float db = red_d[t + off]; int ib = red_i[t + off];
        if (db < red_d[t] || (db == red_d[t] && ib < red_i[t])) { red_d[t] = db; red_i[t] = ib; }
      }
      __syncthreads();
    }
    if (t == 0) {
      const int idx = red_i[0];
      out_tail[s] = (float)idx;
      const float* e = cb + (long)idx * DD;
      float acc = 0.f;
      for (int q = 0; q < DD; ++q) { float df = e[q] - zs[s][q]; acc += df * df; }
      stage_loss[s] = 0.25f * acc / (float)DD;
    }
    __syncthreads();
  }
  if (t == 0) out_tail[SS] = stage_loss[0] + stage_loss[1] + stage_loss[2] + stage_loss[3];

  // bias_total[l][h] = zsum . mod_W[l][:,h] + mod_b[l][h] + dec bias
  for (int e = t; e < LL * HH; e += 256) {
    const int l = e / HH, h = e % HH;
    float acc = mod_b[l * HH + h];
    #pragma unroll 8
    for (int d = 0; d < DD; ++d) acc += zsum[d] * mod_W[((long)l * DD + d) * HH + h];
    acc += (l == 0) ? dec_b0[h] : dec_bh[(long)(l - 1) * HH + h];
    bias_total[e] = acc;
  }
}

// =====================================================================
// Kernel 2a/2b: pack fp32 weights into bf16 WMMA B-fragment layout
// layout element: (((nt*KT + kt)*32 + lane)*16 + j)
//   n = nt*16 + (lane&15) ; k = kt*32 + (lane>>4)*16 + j
// =====================================================================
__global__ void pack_w0(const float* __restrict__ W0, __bf16* __restrict__ W0f)
{
  const int e = blockIdx.x * blockDim.x + threadIdx.x;
  if (e >= 64 * 4 * 32 * 16) return;
  const int j = e & 15, lane = (e >> 4) & 31, kt = (e >> 9) & 3, nt = e >> 11;
  const int n = nt * 16 + (lane & 15);
  const int k = kt * 32 + (lane >> 4) * 16 + j;
  const float v = (k < INR) ? W0[(long)k * HH + n] : 0.0f;
  W0f[e] = (__bf16)v;
}

__global__ void pack_wh(const float* __restrict__ Wh, __bf16* __restrict__ Whf)
{
  const int e = blockIdx.x * blockDim.x + threadIdx.x;
  if (e >= 4 * 64 * 32 * 32 * 16) return;
  const int j = e & 15, lane = (e >> 4) & 31, kt = (e >> 9) & 31;
  const int nt = (e >> 14) & 63, l = e >> 20;
  const int n = nt * 16 + (lane & 15);
  const int k = kt * 32 + (lane >> 4) * 16 + j;
  Whf[e] = (__bf16)Wh[((long)l * HH + k) * HH + n];
}

// =====================================================================
// Kernel 3: fused PE -> 5x modulated WMMA layers (LDS-resident) -> out proj
// 1024 blocks x 256 threads (8 waves). 64 rows/block.
// 2 M-tiles x 2 N-tiles register blocking: each A frag and each B frag
// feeds 2 WMMAs -> 0.5 KB LDS + 0.5 KB L2 traffic per WMMA.
// Dynamic LDS: two 64 x 1032 bf16 activation buffers (264192 B).
// =====================================================================
__global__ __launch_bounds__(256) void mlp_fused(
    const float* __restrict__ coords,
    const __bf16* __restrict__ W0f, const __bf16* __restrict__ Whf,
    const float* __restrict__ bias_total,
    const float* __restrict__ Wout, const float* __restrict__ bout,
    float* __restrict__ out)
{
  extern __shared__ char smem[];
  __bf16* act0 = (__bf16*)smem;               // ROWS*AST bf16
  __bf16* act1 = act0 + ROWS * AST;

  const int t       = threadIdx.x;
  const int lane    = t & 31;
  const int wave    = t >> 5;                 // 0..7
  const int half    = lane >> 4;              // K half selector
  const int lm      = lane & 15;              // row (A) / col (B,C)
  const int rowbase = blockIdx.x * ROWS;

  // ---- positional encoding into act0 (K padded 102 -> 128 with zeros) ----
  {
    const int r = t & 63, part = t >> 6;      // 4 threads per row
    const float c0 = coords[(long)(rowbase + r) * 2 + 0];
    const float c1 = coords[(long)(rowbase + r) * 2 + 1];
    __bf16* row = act0 + r * AST;
    if (part == 0) { row[0] = (__bf16)c0; row[1] = (__bf16)c1; }
    const int f_end = (part * 7 + 7 < FF) ? part * 7 + 7 : FF;
    for (int f = part * 7; f < f_end; ++f) {
      const float w = 3.14159265358979323846f * (float)(1 << f);
      float s0, k0, s1, k1;
      __sincosf(c0 * w, &s0, &k0);
      __sincosf(c1 * w, &s1, &k1);
      const int base = 2 + f * 4;             // (F,2,C) flatten: f*4 + trig*2 + c
      row[base + 0] = (__bf16)s0;
      row[base + 1] = (__bf16)s1;
      row[base + 2] = (__bf16)k0;
      row[base + 3] = (__bf16)k1;
    }
    if (part == 3) for (int c = INR; c < INP; ++c) row[c] = (__bf16)0.0f;
  }
  __syncthreads();

  // ---- 5 modulated GEMM+ReLU layers, activations ping-pong in LDS ----
  __bf16* src = act0;
  __bf16* dst = act1;
  const int mt0    = (wave & 1) * 2;          // M tile pair owned by this wave
  const int mt1    = mt0 + 1;
  const int nphase = wave >> 1;               // 0..3: nt = nphase + 4*i

  for (int l = 0; l < LL; ++l) {
    const int kt_cnt = (l == 0) ? (INP / 32) : (HH / 32);
    const __bf16* W = (l == 0) ? W0f : (Whf + (long)(l - 1) * 64 * 32 * 32 * 16);
    const float* bias = bias_total + l * HH;
    const __bf16* arow0 = src + (mt0 * 16 + lm) * AST;   // lane's A row, tile mt0
    const __bf16* arow1 = src + (mt1 * 16 + lm) * AST;   // lane's A row, tile mt1

    // each wave: 8 iterations of a 2x2 tile block (32 output tiles)
    for (int i = 0; i < 8; ++i) {
      const int nt0 = nphase + 4 * i;         // 0..31
      const int nt1 = nt0 + 32;               // 32..63
      v8f acc00 = {0,0,0,0,0,0,0,0};
      v8f acc01 = {0,0,0,0,0,0,0,0};
      v8f acc10 = {0,0,0,0,0,0,0,0};
      v8f acc11 = {0,0,0,0,0,0,0,0};
      const float bv0 = bias[nt0 * 16 + lm];
      const float bv1 = bias[nt1 * 16 + lm];

      #pragma unroll 2
      for (int kt = 0; kt < kt_cnt; ++kt) {
        FragU a0, a1, b0, b1;
        // A 16x32 bf16: lanes 0-15 hold K[0..15]; lanes 16-31 K[16..31]
        const __bf16* ap0 = arow0 + kt * 32 + half * 16;
        const __bf16* ap1 = arow1 + kt * 32 + half * 16;
        a0.h[0] = *(const float4*)(ap0);
        a0.h[1] = *(const float4*)(ap0 + 8);
        a1.h[0] = *(const float4*)(ap1);
        a1.h[1] = *(const float4*)(ap1 + 8);
        // B 32x16 bf16: per-lane 16 contiguous K, pre-swizzled -> coalesced 32B/lane
        b0.v = *(const bfrag*)(W + (((long)nt0 * kt_cnt + kt) * 32 + lane) * 16);
        b1.v = *(const bfrag*)(W + (((long)nt1 * kt_cnt + kt) * 32 + lane) * 16);
        acc00 = __builtin_amdgcn_wmma_f32_16x16x32_bf16(false, a0.v, false, b0.v,
                                                        (short)0, acc00, false, false);
        acc01 = __builtin_amdgcn_wmma_f32_16x16x32_bf16(false, a0.v, false, b1.v,
                                                        (short)0, acc01, false, false);
        acc10 = __builtin_amdgcn_wmma_f32_16x16x32_bf16(false, a1.v, false, b0.v,
                                                        (short)0, acc10, false, false);
        acc11 = __builtin_amdgcn_wmma_f32_16x16x32_bf16(false, a1.v, false, b1.v,
                                                        (short)0, acc11, false, false);
      }

      // epilogue: +bias(+beta), ReLU, bf16 -> dst LDS
      // C layout: VGPR r = row (half*8 + r), lane&15 = column
      __bf16* d00 = dst + (mt0 * 16 + half * 8) * AST + nt0 * 16 + lm;
      __bf16* d01 = dst + (mt0 * 16 + half * 8) * AST + nt1 * 16 + lm;
      __bf16* d10 = dst + (mt1 * 16 + half * 8) * AST + nt0 * 16 + lm;
      __bf16* d11 = dst + (mt1 * 16 + half * 8) * AST + nt1 * 16 + lm;
      #pragma unroll
      for (int r = 0; r < 8; ++r) {
        float v00 = acc00[r] + bv0; v00 = v00 > 0.f ? v00 : 0.f;
        float v01 = acc01[r] + bv1; v01 = v01 > 0.f ? v01 : 0.f;
        float v10 = acc10[r] + bv0; v10 = v10 > 0.f ? v10 : 0.f;
        float v11 = acc11[r] + bv1; v11 = v11 > 0.f ? v11 : 0.f;
        d00[r * AST] = (__bf16)v00;
        d01[r * AST] = (__bf16)v01;
        d10[r * AST] = (__bf16)v10;
        d11[r * AST] = (__bf16)v11;
      }
    }
    __syncthreads();
    __bf16* tmp = src; src = dst; dst = tmp;
  }
  // after 5 layers (odd count) final h lives in act1 == src; act0 is free.

  // ---- output projection: h(64x1024) @ Wout(1024x3) + bout ----
  {
    const int r = t & 63, q = t >> 6;         // 4 K-slices per row
    const __bf16* hr = src + r * AST + q * 256;
    float a0 = 0.f, a1 = 0.f, a2 = 0.f;
    for (int k = 0; k < 256; ++k) {
      const float hv = (float)hr[k];
      const float* w = Wout + (long)(q * 256 + k) * VV;
      a0 += hv * w[0]; a1 += hv * w[1]; a2 += hv * w[2];
    }
    float* red = (float*)act0;                // 64*4*3 f32 partials (safe alias)
    red[(q * 64 + r) * VV + 0] = a0;
    red[(q * 64 + r) * VV + 1] = a1;
    red[(q * 64 + r) * VV + 2] = a2;
  }
  __syncthreads();
  if (t < 64) {
    const float* red = (const float*)act0;
    #pragma unroll
    for (int v = 0; v < VV; ++v) {
      float s = bout[v];
      for (int q = 0; q < 4; ++q) s += red[(q * 64 + t) * VV + v];
      out[(long)(rowbase + t) * VV + v] = s;
    }
  }
}

// =====================================================================
extern "C" void kernel_launch(void* const* d_in, const int* in_sizes, int n_in,
                              void* d_out, int out_size, void* d_ws, size_t ws_size,
                              hipStream_t stream) {
  const float* coords    = (const float*)d_in[0];
  const int*   lat_idx   = (const int*)  d_in[1];
  const float* latents   = (const float*)d_in[2];
  const float* codebooks = (const float*)d_in[3];
  const float* mod_W     = (const float*)d_in[4];
  const float* mod_b     = (const float*)d_in[5];
  const float* dec_W0    = (const float*)d_in[6];
  const float* dec_b0    = (const float*)d_in[7];
  const float* dec_Wh    = (const float*)d_in[8];
  const float* dec_bh    = (const float*)d_in[9];
  const float* dec_Wout  = (const float*)d_in[10];
  const float* dec_bout  = (const float*)d_in[11];

  char* ws = (char*)d_ws;
  float*  bias_total = (float*) (ws + WS_BIAS_OFF);
  __bf16* W0f        = (__bf16*)(ws + WS_W0_OFF);
  __bf16* Whf        = (__bf16*)(ws + WS_WH_OFF);

  float* out      = (float*)d_out;
  float* out_tail = out + (long)BB * VV;      // idxs (4) + vq loss (1)

  // 1) VQ + fused biases (tiny, single block)
  vq_prep<<<1, 256, 0, stream>>>(latents, lat_idx, codebooks, mod_W, mod_b,
                                 dec_b0, dec_bh, bias_total, out_tail);

  // 2) weight repack fp32 -> bf16 WMMA fragments (~17 MB traffic, ~1 us)
  pack_w0<<<(64 * 4 * 32 * 16 + 255) / 256, 256, 0, stream>>>(dec_W0, W0f);
  pack_wh<<<(4 * 64 * 32 * 32 * 16 + 255) / 256, 256, 0, stream>>>(dec_Wh, Whf);

  // 3) fused MLP: 264 KB dynamic LDS (within CDNA5's 320 KB/WGP)
  const int smem_bytes = 2 * ROWS * AST * (int)sizeof(__bf16);
  hipFuncSetAttribute((const void*)mlp_fused,
                      hipFuncAttributeMaxDynamicSharedMemorySize, smem_bytes);
  mlp_fused<<<BB / ROWS, 256, smem_bytes, stream>>>(
      coords, W0f, Whf, bias_total, dec_Wout, dec_bout, out);
}
